// PoolAggregator_73057393705491
// MI455X (gfx1250) — matrix operand order repaired and checked by reading, not compiled, not run on hardware
//
#include <hip/hip_runtime.h>

typedef float v2f  __attribute__((ext_vector_type(2)));
typedef float v4f  __attribute__((ext_vector_type(4)));
typedef float v8f  __attribute__((ext_vector_type(8)));
typedef __bf16 v16bf __attribute__((ext_vector_type(16)));
typedef unsigned int u32x4 __attribute__((ext_vector_type(4)));
typedef int i32x4 __attribute__((ext_vector_type(4)));
typedef int i32x8 __attribute__((ext_vector_type(8)));

#define BN_EPS 1e-3f

// ---------------------------------------------------------------------------
// TDM: 2D bf16 tile (rows x cols elements, row stride in elements) -> LDS.
// D# built per CDNA5 ISA 8.3/8.4 (group0: count/lds/global/type, group1:
// data_size=2B, tensor dims, tile dims, dim0 stride). 2D => groups 2/3 zero.
// ---------------------------------------------------------------------------
__device__ __forceinline__ void tdm_load_2d_bf16(const __bf16* gsrc,
                                                 unsigned lds_off,
                                                 int rows, int cols,
                                                 int row_stride)
{
#if __has_builtin(__builtin_amdgcn_tensor_load_to_lds)
    unsigned long long ga = (unsigned long long)(uintptr_t)gsrc;
    u32x4 g0;
    g0[0] = 1u;                                        // count=1 (valid user D#)
    g0[1] = lds_off;                                   // lds_addr (bytes)
    g0[2] = (unsigned)(ga & 0xffffffffu);              // global_addr[31:0]
    g0[3] = (unsigned)((ga >> 32) & 0x01ffffffu)       // global_addr[56:32]
          | (2u << 30);                                // type = 2 ("image")
    i32x8 g1;
    g1[0] = (1 << 16);                                 // data_size=1 -> 2 bytes
    g1[1] = (int)((unsigned)(cols & 0xffff) << 16);    // tensor_dim0[15:0]
    g1[2] = (int)((((unsigned)cols >> 16) & 0xffffu)   // tensor_dim0[31:16]
          | ((unsigned)(rows & 0xffff) << 16));        // tensor_dim1[15:0]
    g1[3] = (int)((((unsigned)rows >> 16) & 0xffffu)   // tensor_dim1[31:16]
          | ((unsigned)(cols & 0xffff) << 16));        // tile_dim0
    g1[4] = rows & 0xffff;                             // tile_dim1 (tile_dim2=0)
    g1[5] = row_stride;                                // tensor_dim0_stride[31:0]
    g1[6] = 0;                                         // stride hi / dim1_stride
    g1[7] = 0;
    i32x4 z4 = {0, 0, 0, 0};
#if defined(__clang_major__) && (__clang_major__ >= 23)
    i32x8 z8 = {0, 0, 0, 0, 0, 0, 0, 0};
    __builtin_amdgcn_tensor_load_to_lds(g0, g1, z4, z4, z8, 0);
#else
    __builtin_amdgcn_tensor_load_to_lds(g0, g1, z4, z4, 0);
#endif
#endif
}

// ---------------------------------------------------------------------------
// Prep: split + transpose Wt[256][256] f32 into WtT_hi/WtT_lo bf16 [col][k]
// so B fragments for wmma bf16 are contiguous 16-element runs per lane.
// ---------------------------------------------------------------------------
__global__ __launch_bounds__(256) void pa_split_wt(
    const float* __restrict__ Wt, __bf16* __restrict__ WtT_hi,
    __bf16* __restrict__ WtT_lo)
{
    const int col = blockIdx.x;     // 0..255
    const int k   = threadIdx.x;    // 0..255
    const float x = Wt[(size_t)k * 256 + col];
    const __bf16 h = (__bf16)x;
    WtT_hi[(size_t)col * 256 + k] = h;
    WtT_lo[(size_t)col * 256 + k] = (__bf16)(x - (float)h);
}

// ---------------------------------------------------------------------------
// Kernel 1: h = neigh @ Wt + bt via bf16x3 split WMMA (16x16x32), fused:
//   per-column partial sum/sumsq (BN1 stats) + per-node col max/min (pool).
// Workgroup = 128 rows (4 nodes) x 128 cols; 8 waves, wave w = 16-row block.
// B slab (128 cols x 256 k, hi+lo) staged to LDS by the Tensor Data Mover.
// ---------------------------------------------------------------------------
__global__ __launch_bounds__(256) void pa_gemm1_pool_stats(
    const float*  __restrict__ neigh,    // [M1, 256]
    const __bf16* __restrict__ WtT_hi,   // [256 cols][256 k]
    const __bf16* __restrict__ WtT_lo,
    const float*  __restrict__ bt,       // [256]
    float* __restrict__ pmax,            // [M1/32, 256]
    float* __restrict__ pmin,
    float* __restrict__ psum,            // [gridDim.x, 256]
    float* __restrict__ psq)
{
    __shared__ __bf16 sBhi[128 * 256];   // 64 KB
    __shared__ __bf16 sBlo[128 * 256];   // 64 KB
    __shared__ float  lsum[8][128];
    __shared__ float  lsq [8][128];
    __shared__ float  lmax[8][128];
    __shared__ float  lmin[8][128];

    const int tid  = threadIdx.x;
    const int w    = tid >> 5;
    const int lane = tid & 31;
    const int lm   = lane & 15;
    const int lh   = lane >> 4;
    const int gx   = blockIdx.x;
    const int cy   = blockIdx.y;               // which 128-col half of O
    const int R0   = gx * 128 + w * 16;

    // ---- stage B slab (cols cy*128..cy*128+127, all 256 k) into LDS -------
#if __has_builtin(__builtin_amdgcn_tensor_load_to_lds)
    if (w == 0) {
        tdm_load_2d_bf16(WtT_hi + (size_t)cy * 128 * 256,
                         (unsigned)(uintptr_t)sBhi, 128, 256, 256);
        tdm_load_2d_bf16(WtT_lo + (size_t)cy * 128 * 256,
                         (unsigned)(uintptr_t)sBlo, 128, 256, 256);
        __builtin_amdgcn_s_wait_tensorcnt(0);
    }
#else
    {
        const uint4* srcH = (const uint4*)(WtT_hi + (size_t)cy * 128 * 256);
        const uint4* srcL = (const uint4*)(WtT_lo + (size_t)cy * 128 * 256);
        uint4* dstH = (uint4*)sBhi;
        uint4* dstL = (uint4*)sBlo;
        for (int i = tid; i < 128 * 256 / 8; i += 256) {
            dstH[i] = srcH[i];
            dstL[i] = srcL[i];
        }
    }
#endif
    __syncthreads();

    const float* arow = neigh + (size_t)(R0 + lm) * 256;
    __builtin_prefetch(arow, 0, 3);            // global_prefetch_b8

    v8f acc[8];
#pragma unroll
    for (int f = 0; f < 8; ++f)
        acc[f] = (v8f){0.f, 0.f, 0.f, 0.f, 0.f, 0.f, 0.f, 0.f};

    for (int k0 = 0; k0 < 256; k0 += 32) {
        // A fragment: lanes 0-15 need K = k0+{0..15}; lanes 16-31
        // K = k0+{16..31} (contiguous, matching B). Load f32,
        // split into bf16 hi/lo on the fly.
        const float* aseg0 = arow + k0 + (lh << 4);
        v4f aa[4];
        aa[0] = *(const v4f*)(aseg0);
        aa[1] = *(const v4f*)(aseg0 + 4);
        aa[2] = *(const v4f*)(aseg0 + 8);
        aa[3] = *(const v4f*)(aseg0 + 12);
        v16bf ahi, alo;
#pragma unroll
        for (int i = 0; i < 16; ++i) {
            const float x  = aa[i >> 2][i & 3];
            const __bf16 h = (__bf16)x;
            ahi[i] = h;
            alo[i] = (__bf16)(x - (float)h);
        }
#pragma unroll
        for (int f = 0; f < 8; ++f) {
            // B fragment: col = f*16+lm, K = k0 + lh*16 + 0..15 (contiguous).
            const int boff = (f * 16 + lm) * 256 + k0 + (lh << 4);
            v16bf bhi = *(const v16bf*)(sBhi + boff);
            v16bf blo = *(const v16bf*)(sBlo + boff);
            acc[f] = __builtin_amdgcn_wmma_f32_16x16x32_bf16(
                false, ahi, false, bhi, (short)0, acc[f], false, false);
            acc[f] = __builtin_amdgcn_wmma_f32_16x16x32_bf16(
                false, ahi, false, blo, (short)0, acc[f], false, false);
            acc[f] = __builtin_amdgcn_wmma_f32_16x16x32_bf16(
                false, alo, false, bhi, (short)0, acc[f], false, false);
        }
    }

    // ---- epilogue: bias + per-frag column reductions over the 16 rows -----
#pragma unroll
    for (int f = 0; f < 8; ++f) {
        const int c128 = f * 16 + lm;
        const float bv = bt[cy * 128 + c128];
        float s = 0.f, q = 0.f, mx = -3.402823466e38f, mn = 3.402823466e38f;
#pragma unroll
        for (int j = 0; j < 8; ++j) {
            const float v = acc[f][j] + bv;    // rows: lh==0 -> M=j, else M=8+j
            s += v; q += v * v;
            mx = fmaxf(mx, v); mn = fminf(mn, v);
        }
        s += __shfl_xor(s, 16, 32);
        q += __shfl_xor(q, 16, 32);
        mx = fmaxf(mx, __shfl_xor(mx, 16, 32));
        mn = fminf(mn, __shfl_xor(mn, 16, 32));
        if (lh == 0) {
            lsum[w][c128] = s;  lsq [w][c128] = q;
            lmax[w][c128] = mx; lmin[w][c128] = mn;
        }
    }
    __syncthreads();

    // Stats: sum across all 8 row-blocks (threads 0..127, one column each).
    if (tid < 128) {
        float s = 0.f, q = 0.f;
#pragma unroll
        for (int ww = 0; ww < 8; ++ww) { s += lsum[ww][tid]; q += lsq[ww][tid]; }
        psum[(size_t)gx * 256 + cy * 128 + tid] = s;
        psq [(size_t)gx * 256 + cy * 128 + tid] = q;
    }
    // Pool: node n (rows 32n..32n+31) = waves 2n, 2n+1. 4 nodes x 128 cols.
    {
        const int c  = tid & 127;
        const int nb = tid >> 7;                 // 0 or 1
#pragma unroll
        for (int it = 0; it < 2; ++it) {
            const int node = nb + 2 * it;
            const int wn = node * 2;
            pmax[(size_t)(gx * 4 + node) * 256 + cy * 128 + c] =
                fmaxf(lmax[wn][c], lmax[wn + 1][c]);
            pmin[(size_t)(gx * 4 + node) * 256 + cy * 128 + c] =
                fminf(lmin[wn][c], lmin[wn + 1][c]);
        }
    }
}

// ---------------------------------------------------------------------------
// Column stats reduction -> BN affine (scale, shift). Deterministic tree.
// ---------------------------------------------------------------------------
__global__ __launch_bounds__(256) void pa_colstats(
    const float* __restrict__ psum, const float* __restrict__ psq,
    const float* __restrict__ gamma, const float* __restrict__ beta,
    float* __restrict__ scale, float* __restrict__ shift,
    int count, int stride, float invN)
{
    __shared__ float rs[256], rq[256];
    const int col = blockIdx.x;
    float s = 0.f, q = 0.f;
    for (int i = threadIdx.x; i < count; i += 256) {
        s += psum[(size_t)i * stride + col];
        q += psq [(size_t)i * stride + col];
    }
    rs[threadIdx.x] = s; rq[threadIdx.x] = q;
    __syncthreads();
    for (int off = 128; off > 0; off >>= 1) {
        if (threadIdx.x < off) {
            rs[threadIdx.x] += rs[threadIdx.x + off];
            rq[threadIdx.x] += rq[threadIdx.x + off];
        }
        __syncthreads();
    }
    if (threadIdx.x == 0) {
        const float mu  = rs[0] * invN;
        const float var = rq[0] * invN - mu * mu;
        const float sc  = gamma[col] * rsqrtf(var + BN_EPS);
        scale[col] = sc;
        shift[col] = beta[col] - mu * sc;
    }
}

// ---------------------------------------------------------------------------
// pooled_bn = scale>=0 ? scale*max + shift : scale*min + shift
// ---------------------------------------------------------------------------
__global__ void pa_apply_bn_pool(
    const float* __restrict__ pmax, const float* __restrict__ pmin,
    const float* __restrict__ scale, const float* __restrict__ shift,
    float* __restrict__ pooled, int total)
{
    const int i = blockIdx.x * 256 + threadIdx.x;
    if (i >= total) return;
    const int o = i & 255;
    const float sc = scale[o];
    const float v  = (sc >= 0.f) ? pmax[i] : pmin[i];
    pooled[i] = sc * v + shift[o];
}

// ---------------------------------------------------------------------------
// Kernel 3: raw[:,0:256] = self @ W_self ; raw[:,256:512] = pooled_bn @ W_neigh
// fp32 WMMA 16x16x4 (only ~5 GFLOP); fused BN2 stat partials; row masking.
// ---------------------------------------------------------------------------
__global__ __launch_bounds__(256) void pa_gemm2_stats(
    const float* __restrict__ A0, const float* __restrict__ A1,
    const float* __restrict__ W0, const float* __restrict__ W1,
    float* __restrict__ raw, float* __restrict__ psum, float* __restrict__ psq,
    int M)
{
    __shared__ float lsum[8][128];
    __shared__ float lsq [8][128];

    const int tid  = threadIdx.x;
    const int w    = tid >> 5;
    const int lane = tid & 31;
    const int rb   = w & 3;
    const int cw   = w >> 2;
    const int lm   = lane & 15;
    const int lh   = lane >> 4;
    const int half = blockIdx.y;
    const int R0   = blockIdx.x * 64 + rb * 16;
    const int colb = cw * 128;
    const int koff = lh << 1;

    const float* A = half ? A1 : A0;
    const float* W = half ? W1 : W0;

    int row = R0 + lm;
    if (row >= M) row = M - 1;                 // clamp: keep EXEC full for WMMA
    const float* arow = A + (size_t)row * 256;

    v8f acc[8];
#pragma unroll
    for (int f = 0; f < 8; ++f)
        acc[f] = (v8f){0.f, 0.f, 0.f, 0.f, 0.f, 0.f, 0.f, 0.f};

#pragma unroll 2
    for (int k0 = 0; k0 < 256; k0 += 4) {
        const int kb = k0 + koff;
        v2f a = *(const v2f*)(arow + kb);
#pragma unroll
        for (int f = 0; f < 8; ++f) {
            const int col = colb + f * 16 + lm;
            v2f b;
            b.x = W[(size_t)kb * 256 + col];
            b.y = W[(size_t)(kb + 1) * 256 + col];
            acc[f] = __builtin_amdgcn_wmma_f32_16x16x4_f32(
                false, a, false, b, (short)0, acc[f], false, false);
        }
    }

#pragma unroll
    for (int f = 0; f < 8; ++f) {
        const int c128 = f * 16 + lm;
        const int gcol = half * 256 + colb + c128;
        float s = 0.f, q = 0.f;
#pragma unroll
        for (int j = 0; j < 8; ++j) {
            const int lrow = R0 + (lh << 3) + j;
            const float v  = acc[f][j];
            if (lrow < M) {
                s += v; q += v * v;
                raw[(size_t)lrow * 512 + gcol] = v;
            }
        }
        s += __shfl_xor(s, 16, 32);
        q += __shfl_xor(q, 16, 32);
        if (lh == 0) { lsum[w][c128] = s; lsq[w][c128] = q; }
    }
    __syncthreads();

    const int col256 = tid;
    const int cwv = col256 >> 7, cc = col256 & 127, wb = cwv * 4;
    float s = lsum[wb+0][cc] + lsum[wb+1][cc] + lsum[wb+2][cc] + lsum[wb+3][cc];
    float q = lsq [wb+0][cc] + lsq [wb+1][cc] + lsq [wb+2][cc] + lsq [wb+3][cc];
    psum[(size_t)blockIdx.x * 512 + half * 256 + col256] = s;
    psq [(size_t)blockIdx.x * 512 + half * 256 + col256] = q;
}

// ---------------------------------------------------------------------------
__global__ void pa_bn2_relu(
    const float* __restrict__ raw,
    const float* __restrict__ scale, const float* __restrict__ shift,
    float* __restrict__ out, int total)
{
    const int i = blockIdx.x * 256 + threadIdx.x;
    if (i >= total) return;
    const int c = i & 511;
    const float v = raw[i] * scale[c] + shift[c];
    out[i] = v > 0.f ? v : 0.f;
}

// ---------------------------------------------------------------------------
extern "C" void kernel_launch(void* const* d_in, const int* in_sizes, int n_in,
                              void* d_out, int out_size, void* d_ws, size_t ws_size,
                              hipStream_t stream)
{
    const float* self_nodes = (const float*)d_in[0];
    const float* neigh      = (const float*)d_in[1];
    /* d_in[2] = len_adj_nodes: unused by the reference */
    const float* Wt         = (const float*)d_in[3];
    const float* bt         = (const float*)d_in[4];
    const float* gamma1     = (const float*)d_in[5];
    const float* beta1      = (const float*)d_in[6];
    const float* W_self     = (const float*)d_in[7];
    const float* W_neigh    = (const float*)d_in[8];
    const float* gamma2     = (const float*)d_in[9];
    const float* beta2      = (const float*)d_in[10];

    const int D  = 256, O = 256;
    const int N  = in_sizes[0] / D;           // 20000
    const int K  = in_sizes[1] / (N * D);     // 32
    const int M1 = N * K;                     // 640000
    const int G1 = M1 / 128;                  // 5000 (128 rows per WG)
    const int G3 = (N + 63) / 64;             // 313

    float* ws = (float*)d_ws;
    __bf16* wt_hi = (__bf16*)ws;  ws += (size_t)D * O / 2;   // 65536 bf16
    __bf16* wt_lo = (__bf16*)ws;  ws += (size_t)D * O / 2;
    float* pmax   = ws;  ws += (size_t)N * O;
    float* pmin   = ws;  ws += (size_t)N * O;
    float* pooled = ws;  ws += (size_t)N * O;
    float* psum1  = ws;  ws += (size_t)G1 * O;
    float* psq1   = ws;  ws += (size_t)G1 * O;
    float* scale1 = ws;  ws += O;
    float* shift1 = ws;  ws += O;
    float* raw2   = ws;  ws += (size_t)N * 2 * O;
    float* psum2  = ws;  ws += (size_t)G3 * 2 * O;
    float* psq2   = ws;  ws += (size_t)G3 * 2 * O;
    float* scale2 = ws;  ws += 2 * O;
    float* shift2 = ws;  ws += 2 * O;

    pa_split_wt<<<O, 256, 0, stream>>>(Wt, wt_hi, wt_lo);
    pa_gemm1_pool_stats<<<dim3(G1, 2), 256, 0, stream>>>(neigh, wt_hi, wt_lo, bt,
                                                         pmax, pmin, psum1, psq1);
    pa_colstats<<<O, 256, 0, stream>>>(psum1, psq1, gamma1, beta1, scale1, shift1,
                                       G1, O, 1.0f / (float)M1);
    pa_apply_bn_pool<<<(N * O + 255) / 256, 256, 0, stream>>>(pmax, pmin, scale1, shift1,
                                                              pooled, N * O);
    pa_gemm2_stats<<<dim3(G3, 2), 256, 0, stream>>>(self_nodes, pooled, W_self, W_neigh,
                                                    raw2, psum2, psq2, N);
    pa_colstats<<<2 * O, 256, 0, stream>>>(psum2, psq2, gamma2, beta2, scale2, shift2,
                                           G3, 2 * O, 1.0f / (float)N);
    pa_bn2_relu<<<(N * 2 * O + 255) / 256, 256, 0, stream>>>(raw2, scale2, shift2,
                                                             (float*)d_out, N * 2 * O);
}
